// Model_15814069584221
// MI455X (gfx1250) — matrix operand order repaired and checked by reading, not compiled, not run on hardware
//
#include <hip/hip_runtime.h>
#include <hip/hip_bf16.h>
#include <cstring>

// ---------------- model dims ----------------
#define DS   128   // n_seq S
#define DR   256   // n_res R
#define DCM  256   // C_M
#define DCZ  128   // C_Z
#define HMSA 8
#define HTRI 4
#define CHD  32    // C_HEAD
#define COPM 32
#define CTRI 128
#define NLAYERS 2
#define ATT_SCALE 0.1767766952966369f   // 1/sqrt(32)

typedef __bf16 bf16;
typedef bf16  bf16x8 __attribute__((ext_vector_type(8)));
typedef bf16  v16bf  __attribute__((ext_vector_type(16)));
typedef float v8f    __attribute__((ext_vector_type(8)));

// ---------------- generalized batched GEMM ----------------
// C[b1,b2,m,n] (+)= gate * act( alpha * sum_k A(m,k) * B(k,n) + bias )
// Requirements (all call sites satisfy them): M % 64 == 0, K % 32 == 0,
// all *div / b2n are powers of two; on non-contiguous ("slow") operand paths kdiv == 1.
struct GemmP {
  const float* A; const float* B; float* C;
  const float* bias; const float* gate;
  int M, N, K, b2n;
  long a_m1, a_m2, a_k1, a_k2, a_b1, a_b2; int a_mdiv, a_kdiv;
  long b_k1, b_k2, b_n1, b_n2, b_b1, b_b2; int b_kdiv, b_ndiv;
  long c_m, c_n, c_b1, c_b2;
  long s_m, s_n, s_b1, s_b2;       // bias strides
  float alpha;
  int act;        // 0 none, 1 relu, 2 sigmoid
  int residual;   // 1 -> C += val
  // ---- derived (filled by launch_gemm) ----
  int a_msh, a_mmask, a_ksh, a_kmask;
  int b_ksh, b_kmask, b_nsh, b_nmask;
  int b2sh, a_fast, b_fast;
};

__global__ __launch_bounds__(256) void gemm_bf16_wmma(GemmP p) {
  __shared__ __align__(16) bf16 As[64][32];    // [m][k]
  __shared__ __align__(16) bf16 Bs[128][32];   // [n][k] (k-contiguous per column)
  const int tid  = threadIdx.x;
  const int lane = tid & 31;
  const int wave = tid >> 5;
  const int wm = wave & 3;          // 4 wave rows -> 16 M each
  const int wn = wave >> 2;         // 2 wave cols -> 64 N each
  const int z  = blockIdx.z;
  const int b1 = z >> p.b2sh, b2 = z & ((1 << p.b2sh) - 1);
  const int m0 = blockIdx.y * 64, n0 = blockIdx.x * 128;

  // ---- staging thread mapping ----
  const int am_  = m0 + (tid >> 2);       // A: each thread stages 8 k of one m-row
  const int askg = (tid & 3) * 8;
  const int bn_  = tid >> 1;              // B: each thread stages 16 k of one n-col
  const int bskg = (tid & 1) * 16;
  const int bng  = n0 + bn_;
  const bool bvalid = bng < p.N;

  long abase = ((long)(am_ >> p.a_msh)) * p.a_m1 + (long)(am_ & p.a_mmask) * p.a_m2
             + ((long)(askg >> p.a_ksh)) * p.a_k1 + (long)(askg & p.a_kmask) * p.a_k2
             + (long)b1 * p.a_b1 + (long)b2 * p.a_b2;
  long bbase = ((long)(bskg >> p.b_ksh)) * p.b_k1 + (long)(bskg & p.b_kmask) * p.b_k2
             + ((long)(bng >> p.b_nsh)) * p.b_n1 + (long)(bng & p.b_nmask) * p.b_n2
             + (long)b1 * p.b_b1 + (long)b2 * p.b_b2;
  const long astep = (long)(32 >> p.a_ksh) * p.a_k1;
  const long bstep = (long)(32 >> p.b_ksh) * p.b_k1;

  float ar[8];
  float br[16];
  auto loadTiles = [&]() {
    if (p.a_fast) {
      const float4 x0 = *(const float4*)(p.A + abase);
      const float4 x1 = *(const float4*)(p.A + abase + 4);
      ar[0] = x0.x; ar[1] = x0.y; ar[2] = x0.z; ar[3] = x0.w;
      ar[4] = x1.x; ar[5] = x1.y; ar[6] = x1.z; ar[7] = x1.w;
    } else {
      #pragma unroll
      for (int j = 0; j < 8; ++j) ar[j] = p.A[abase + (long)j * p.a_k1];   // slow path has a_ksh==0
    }
    if (bvalid) {
      if (p.b_fast) {
        #pragma unroll
        for (int q = 0; q < 4; ++q) {
          const float4 x = *(const float4*)(p.B + bbase + 4 * q);
          br[4 * q + 0] = x.x; br[4 * q + 1] = x.y; br[4 * q + 2] = x.z; br[4 * q + 3] = x.w;
        }
      } else {
        #pragma unroll
        for (int j = 0; j < 16; ++j) br[j] = p.B[bbase + (long)j * p.b_k1];
      }
    } else {
      #pragma unroll
      for (int j = 0; j < 16; ++j) br[j] = 0.0f;
    }
  };
  auto storeTiles = [&]() {
    bf16x8 va;
    #pragma unroll
    for (int j = 0; j < 8; ++j) va[j] = (bf16)ar[j];
    *(bf16x8*)&As[tid >> 2][askg] = va;
    bf16x8 vb0, vb1;
    #pragma unroll
    for (int j = 0; j < 8; ++j) { vb0[j] = (bf16)br[j]; vb1[j] = (bf16)br[j + 8]; }
    *(bf16x8*)&Bs[bn_][bskg] = vb0;
    *(bf16x8*)&Bs[bn_][bskg + 8] = vb1;
  };

  v8f acc[4] = {};
  const int nk = p.K >> 5;

  loadTiles();
  storeTiles();
  for (int kt = 0; kt < nk; ++kt) {
    __syncthreads();                      // staged tile visible
    const bool more = (kt + 1) < nk;
    abase += astep; bbase += bstep;
    if (more) loadTiles();                // overlap next-tile loads with WMMA
    // A fragment 16x32: lanes 0-15 -> K {0..7,16..23}; lanes 16-31 -> {8..15,24..31}
    const int amr = wm * 16 + (lane & 15);
    const int akb = (lane >> 4) * 8;
    union { v16bf v; bf16x8 h[2]; } af;
    af.h[0] = *(const bf16x8*)&As[amr][akb];
    af.h[1] = *(const bf16x8*)&As[amr][akb + 16];
    // B fragment 32x16: lanes 0-15 -> K {0..7,16..23}; lanes 16-31 -> {8..15,24..31}
    const int bnr = wn * 64 + (lane & 15);
    const int bkb = (lane >> 4) * 8;
    #pragma unroll
    for (int t = 0; t < 4; ++t) {
      union { v16bf v; bf16x8 h[2]; } bfr;
      bfr.h[0] = *(const bf16x8*)&Bs[bnr + t * 16][bkb];
      bfr.h[1] = *(const bf16x8*)&Bs[bnr + t * 16][bkb + 16];
      acc[t] = __builtin_amdgcn_wmma_f32_16x16x32_bf16(false, af.v, false, bfr.v, (short)0, acc[t], false, false);
    }
    __syncthreads();                      // all reads done before overwrite
    if (more) storeTiles();
  }

  // ---- epilogue ----
  const long cB = (long)b1 * p.c_b1 + (long)b2 * p.c_b2;
  const long sB = (long)b1 * p.s_b1 + (long)b2 * p.s_b2;
  #pragma unroll
  for (int t = 0; t < 4; ++t) {
    const int n = n0 + wn * 64 + t * 16 + (lane & 15);
    if (n >= p.N) continue;
    #pragma unroll
    for (int r = 0; r < 8; ++r) {
      const int m = m0 + wm * 16 + r + 8 * (lane >> 4);
      float v = acc[t][r] * p.alpha;
      if (p.bias) v += p.bias[(long)m * p.s_m + (long)n * p.s_n + sB];
      if (p.act == 1)      v = fmaxf(v, 0.0f);
      else if (p.act == 2) v = 1.0f / (1.0f + __expf(-v));
      const long co = (long)m * p.c_m + (long)n * p.c_n + cB;
      if (p.gate) v *= p.gate[co];
      if (p.residual) p.C[co] += v; else p.C[co] = v;
    }
  }
}

// ---------------- LayerNorm: one wave32 per row ----------------
__global__ __launch_bounds__(256) void ln_kernel(const float* __restrict__ x,
                                                 const float* __restrict__ g,
                                                 const float* __restrict__ b,
                                                 float* __restrict__ y, int rows, int C) {
  const int wave = threadIdx.x >> 5, lane = threadIdx.x & 31;
  const int row = blockIdx.x * 8 + wave;
  if (row >= rows) return;
  const float* xr = x + (long)row * C;
  float s = 0.0f, sq = 0.0f;
  for (int c = lane; c < C; c += 32) { float v = xr[c]; s += v; sq += v * v; }
  for (int o = 16; o; o >>= 1) { s += __shfl_xor(s, o, 32); sq += __shfl_xor(sq, o, 32); }
  const float mu = s / C;
  const float rs = rsqrtf(sq / C - mu * mu + 1e-5f);
  float* yr = y + (long)row * C;
  for (int c = lane; c < C; c += 32) yr[c] = (xr[c] - mu) * rs * g[c] + b[c];
}

// ---------------- softmax over last dim: one wave32 per row ----------------
__global__ __launch_bounds__(256) void softmax_kernel(float* __restrict__ x, int rows, int L) {
  const int wave = threadIdx.x >> 5, lane = threadIdx.x & 31;
  const int row = blockIdx.x * 8 + wave;
  if (row >= rows) return;
  float* xr = x + (long)row * L;
  float mx = -1e30f;
  for (int j = lane; j < L; j += 32) mx = fmaxf(mx, xr[j]);
  for (int o = 16; o; o >>= 1) mx = fmaxf(mx, __shfl_xor(mx, o, 32));
  float s = 0.0f;
  for (int j = lane; j < L; j += 32) { float e = __expf(xr[j] - mx); xr[j] = e; s += e; }
  for (int o = 16; o; o >>= 1) s += __shfl_xor(s, o, 32);
  const float inv = 1.0f / s;
  for (int j = lane; j < L; j += 32) xr[j] *= inv;
}

// ---------------- host helpers ----------------
static inline GemmP plainAB(const float* A, const float* B, float* C, int M, int N, int K) {
  GemmP p; memset(&p, 0, sizeof(p));
  p.A = A; p.B = B; p.C = C;
  p.M = M; p.N = N; p.K = K; p.b2n = 1;
  p.a_mdiv = 1; p.a_kdiv = 1; p.b_kdiv = 1; p.b_ndiv = 1;
  p.a_m1 = K; p.a_k1 = 1;          // A row-major [M][K]
  p.b_k1 = N; p.b_n1 = 1;          // B row-major [K][N]
  p.c_m = N; p.c_n = 1;
  p.s_n = 1;
  p.alpha = 1.0f;
  return p;
}
static inline int ilog2(int x) { int s = 0; while ((1 << s) < x) ++s; return s; }
static inline void launch_gemm(hipStream_t st, const GemmP& pin, int b1n = 1) {
  GemmP p = pin;
  p.a_msh = ilog2(p.a_mdiv); p.a_mmask = p.a_mdiv - 1;
  p.a_ksh = ilog2(p.a_kdiv); p.a_kmask = p.a_kdiv - 1;
  p.b_ksh = ilog2(p.b_kdiv); p.b_kmask = p.b_kdiv - 1;
  p.b_nsh = ilog2(p.b_ndiv); p.b_nmask = p.b_ndiv - 1;
  p.b2sh  = ilog2(p.b2n);
  p.a_fast = ((p.a_ksh == 0 && p.a_k1 == 1) || (p.a_ksh >= 3 && p.a_k2 == 1)) ? 1 : 0;
  p.b_fast = ((p.b_ksh == 0 && p.b_k1 == 1) || (p.b_ksh >= 4 && p.b_k2 == 1)) ? 1 : 0;
  dim3 g((p.N + 127) / 128, (p.M + 63) / 64, (unsigned)(b1n * p.b2n));
  hipLaunchKernelGGL(gemm_bf16_wmma, g, dim3(256), 0, st, p);
}
static inline void launch_ln(hipStream_t st, const float* x, const float* g, const float* b,
                             float* y, int rows, int C) {
  hipLaunchKernelGGL(ln_kernel, dim3((rows + 7) / 8), dim3(256), 0, st, x, g, b, y, rows, C);
}
static inline void launch_softmax(hipStream_t st, float* x, int rows, int L) {
  hipLaunchKernelGGL(softmax_kernel, dim3((rows + 7) / 8), dim3(256), 0, st, x, rows, L);
}

// ---------------- orchestration ----------------
// Input flattening assumption (JAX pytree sorted-key order):
//   0:msa 1:pair 2:Wms 3:bms [4 + 79*layer ...] 162:norm_pair_b 163:norm_pair_g
// Per layer (sorted sub-dicts): col(9) @0, msa_tr(6) @9, opm(6) @15, pair_tr(6) @21,
//   row(12) @27, tri_end(10) @39, tri_in(10) @49, tri_out(10) @59, tri_start(10) @69.
extern "C" void kernel_launch(void* const* d_in, const int* in_sizes, int n_in,
                              void* d_out, int out_size, void* d_ws, size_t ws_size,
                              hipStream_t stream) {
  (void)in_sizes; (void)n_in; (void)out_size; (void)ws_size;
  const float* const* in = (const float* const*)d_in;
  float* out = (float*)d_out;

  const long MSA_N  = (long)DS * DR * DCM;   // 8388608
  const long PAIR_N = (long)DR * DR * DCZ;   // 8388608

  float* msa     = out;                // working msa lives in its output slot
  float* pairOut = out + MSA_N;        // final LN'd pair (scratch until the end)
  float* msas    = out + 2 * MSA_N;

  const long SLAB = 8388608;           // 33.5 MB slabs; all chunk sizes == SLAB
  float* W   = (float*)d_ws;
  float* zp  = W + 0 * SLAB;  // working pair
  float* wln = W + 1 * SLAB;  // LN output
  float* wq  = W + 2 * SLAB;
  float* wk  = W + 3 * SLAB;
  float* wv  = W + 4 * SLAB;
  float* wg  = W + 5 * SLAB;
  float* wo  = W + 6 * SLAB;  // attn output / abln
  float* wt  = W + 7 * SLAB;  // logits / hidden / OPM chunk
  float* wx  = pairOut;       // pair-bias & small scratch; overwritten by final LN

  hipMemcpyAsync(msa, in[0], MSA_N * sizeof(float), hipMemcpyDeviceToDevice, stream);
  hipMemcpyAsync(zp,  in[1], PAIR_N * sizeof(float), hipMemcpyDeviceToDevice, stream);

  for (int ly = 0; ly < NLAYERS; ++ly) {
    const int base = 4 + 79 * ly;
    auto P = [&](int i) -> const float* { return in[base + i]; };
    GemmP p;

    // ================= MSA row attention (pair-biased, gated) =================
    {
      const float *Wb=P(27), *Wg_=P(28), *Wk_=P(29), *Wo_=P(30), *Wq_=P(31), *Wv_=P(32),
                  *bg=P(33), *bo=P(34), *ln_b=P(35), *ln_g=P(36), *lnz_b=P(37), *lnz_g=P(38);
      launch_ln(stream, msa, ln_g, ln_b, wln, DS * DR, DCM);
      launch_ln(stream, zp, lnz_g, lnz_b, wo, DR * DR, DCZ);           // zln -> wo (temp)
      p = plainAB(wln, Wq_, wq, DS * DR, DCM, DCM); launch_gemm(stream, p);
      p = plainAB(wln, Wk_, wk, DS * DR, DCM, DCM); launch_gemm(stream, p);
      p = plainAB(wln, Wv_, wv, DS * DR, DCM, DCM); launch_gemm(stream, p);
      p = plainAB(wln, Wg_, wg, DS * DR, DCM, DCM); p.bias = bg; p.act = 2; launch_gemm(stream, p);
      p = plainAB(wo, Wb, wx, DR * DR, HMSA, DCZ); launch_gemm(stream, p);  // pair bias [R,R,8]
      for (int sc = 0; sc < 8; ++sc) {
        const long s0 = (long)sc * 16;
        // logits[s,h,i,j] = scale * q.k + bias[h,i,j]
        p = plainAB(wq + s0 * DR * DCM, wk + s0 * DR * DCM, wt, DR, DR, CHD);
        p.a_m1 = DCM; p.a_k1 = 1; p.a_b1 = (long)DR * DCM; p.a_b2 = CHD;
        p.b_k1 = 1; p.b_n1 = DCM; p.b_b1 = (long)DR * DCM; p.b_b2 = CHD;
        p.c_m = DR; p.c_n = 1; p.c_b1 = (long)HMSA * DR * DR; p.c_b2 = (long)DR * DR;
        p.alpha = ATT_SCALE;
        p.bias = wx; p.s_m = (long)DR * HMSA; p.s_n = HMSA; p.s_b1 = 0; p.s_b2 = 1;
        p.b2n = HMSA;
        launch_gemm(stream, p, 16);
        launch_softmax(stream, wt, 16 * HMSA * DR, DR);
        // o = (softmax @ V) * gate
        p = plainAB(wt, wv + s0 * DR * DCM, wo + s0 * DR * DCM, DR, CHD, DR);
        p.a_m1 = DR; p.a_k1 = 1; p.a_b1 = (long)HMSA * DR * DR; p.a_b2 = (long)DR * DR;
        p.b_k1 = DCM; p.b_n1 = 1; p.b_b1 = (long)DR * DCM; p.b_b2 = CHD;
        p.c_m = DCM; p.c_n = 1; p.c_b1 = (long)DR * DCM; p.c_b2 = CHD;
        p.gate = wg + s0 * DR * DCM;
        p.b2n = HMSA;
        launch_gemm(stream, p, 16);
      }
      p = plainAB(wo, Wo_, msa, DS * DR, DCM, DCM); p.bias = bo; p.residual = 1; launch_gemm(stream, p);
    }

    // ================= MSA column attention (gated) =================
    {
      const float *Wg_=P(0), *Wk_=P(1), *Wo_=P(2), *Wq_=P(3), *Wv_=P(4),
                  *bg=P(5), *bo=P(6), *ln_b=P(7), *ln_g=P(8);
      launch_ln(stream, msa, ln_g, ln_b, wln, DS * DR, DCM);
      p = plainAB(wln, Wq_, wq, DS * DR, DCM, DCM); launch_gemm(stream, p);
      p = plainAB(wln, Wk_, wk, DS * DR, DCM, DCM); launch_gemm(stream, p);
      p = plainAB(wln, Wv_, wv, DS * DR, DCM, DCM); launch_gemm(stream, p);
      p = plainAB(wln, Wg_, wg, DS * DR, DCM, DCM); p.bias = bg; p.act = 2; launch_gemm(stream, p);
      for (int rc = 0; rc < 4; ++rc) {
        const long r0 = (long)rc * 64;
        p = plainAB(wq + r0 * DCM, wk + r0 * DCM, wt, DS, DS, CHD);
        p.a_m1 = (long)DR * DCM; p.a_k1 = 1; p.a_b1 = DCM; p.a_b2 = CHD;
        p.b_k1 = 1; p.b_n1 = (long)DR * DCM; p.b_b1 = DCM; p.b_b2 = CHD;
        p.c_m = DS; p.c_n = 1; p.c_b1 = (long)HMSA * DS * DS; p.c_b2 = (long)DS * DS;
        p.alpha = ATT_SCALE; p.b2n = HMSA;
        launch_gemm(stream, p, 64);
        launch_softmax(stream, wt, 64 * HMSA * DS, DS);
        p = plainAB(wt, wv + r0 * DCM, wo + r0 * DCM, DS, CHD, DS);
        p.a_m1 = DS; p.a_k1 = 1; p.a_b1 = (long)HMSA * DS * DS; p.a_b2 = (long)DS * DS;
        p.b_k1 = (long)DR * DCM; p.b_n1 = 1; p.b_b1 = DCM; p.b_b2 = CHD;
        p.c_m = (long)DR * DCM; p.c_n = 1; p.c_b1 = DCM; p.c_b2 = CHD;
        p.gate = wg + r0 * DCM; p.b2n = HMSA;
        launch_gemm(stream, p, 64);
      }
      p = plainAB(wo, Wo_, msa, DS * DR, DCM, DCM); p.bias = bo; p.residual = 1; launch_gemm(stream, p);
    }

    // ================= MSA transition =================
    {
      const float *W1=P(9), *W2=P(10), *b1=P(11), *b2=P(12), *ln_b=P(13), *ln_g=P(14);
      launch_ln(stream, msa, ln_g, ln_b, wln, DS * DR, DCM);
      for (int c = 0; c < 4; ++c) {
        const long r0 = (long)c * 8192;
        p = plainAB(wln + r0 * DCM, W1, wt, 8192, 4 * DCM, DCM); p.bias = b1; p.act = 1; launch_gemm(stream, p);
        p = plainAB(wt, W2, msa + r0 * DCM, 8192, DCM, 4 * DCM); p.bias = b2; p.residual = 1; launch_gemm(stream, p);
      }
    }

    // ================= Outer product mean (replaces pair) =================
    {
      const float *Wa=P(15), *Wb=P(16), *Wo_=P(17), *bo=P(18), *ln_b=P(19), *ln_g=P(20);
      launch_ln(stream, msa, ln_g, ln_b, wln, DS * DR, DCM);
      p = plainAB(wln, Wa, wq, DS * DR, COPM, DCM); launch_gemm(stream, p);   // a [S,R,32]
      p = plainAB(wln, Wb, wk, DS * DR, COPM, DCM); launch_gemm(stream, p);   // b [S,R,32]
      const long RC = (long)DR * COPM;  // 8192
      for (int ic = 0; ic < 8; ++ic) {
        const long I0 = (long)ic * 32;
        // o[(ii,c),(j,d)] = (1/S) sum_s a[s,I0+ii,c] b[s,j,d]
        p = plainAB(wq + I0 * COPM, wk, wt, 32 * COPM, DR * COPM, DS);
        p.a_mdiv = COPM; p.a_m1 = COPM; p.a_m2 = 1; p.a_k1 = RC; p.a_k2 = 0;
        p.b_k1 = RC; p.b_n1 = 1;
        p.c_m = RC; p.c_n = 1;
        p.alpha = 1.0f / DS;
        launch_gemm(stream, p);
        // pair[(I0+ii)*R + j, :] = o.reshape(...,1024) @ Wo + bo
        p = plainAB(wt, Wo_, zp + I0 * DR * DCZ, 32 * DR, DCZ, COPM * COPM);
        p.a_mdiv = DR; p.a_m1 = (long)COPM * RC; p.a_m2 = COPM;
        p.a_kdiv = COPM; p.a_k1 = RC; p.a_k2 = 1;
        p.bias = bo;
        launch_gemm(stream, p);
      }
    }

    // ================= Triangle multiplications =================
    auto tri_mul = [&](int pb, bool outgoing) {
      const float *Wa=P(pb+0), *Wag=P(pb+1), *Wb=P(pb+2), *Wbg=P(pb+3), *Wg_=P(pb+4), *Wo_=P(pb+5),
                  *ln2_b=P(pb+6), *ln2_g=P(pb+7), *ln_b=P(pb+8), *ln_g=P(pb+9);
      GemmP q;
      launch_ln(stream, zp, ln_g, ln_b, wln, DR * DR, DCZ);
      q = plainAB(wln, Wa, wq, DR * DR, CTRI, DCZ); launch_gemm(stream, q);
      q = plainAB(wln, Wag, wq, DR * DR, CTRI, DCZ); q.act = 2; q.gate = wq; launch_gemm(stream, q); // a = sig(zWag)*zWa
      q = plainAB(wln, Wb, wk, DR * DR, CTRI, DCZ); launch_gemm(stream, q);
      q = plainAB(wln, Wbg, wk, DR * DR, CTRI, DCZ); q.act = 2; q.gate = wk; launch_gemm(stream, q);
      const long RZ = (long)DR * CTRI;  // 32768
      // ab[i,j,c], batched over c
      q = plainAB(wq, wk, wv, DR, DR, DR);
      if (outgoing) { q.a_m1 = RZ; q.a_k1 = CTRI; q.b_k1 = CTRI; q.b_n1 = RZ; }
      else          { q.a_m1 = CTRI; q.a_k1 = RZ; q.b_k1 = RZ; q.b_n1 = CTRI; }
      q.a_b1 = 1; q.b_b1 = 1;
      q.c_m = RZ; q.c_n = CTRI; q.c_b1 = 1;
      launch_gemm(stream, q, CTRI);
      launch_ln(stream, wv, ln2_g, ln2_b, wo, DR * DR, CTRI);
      q = plainAB(wln, Wg_, wg, DR * DR, DCZ, DCZ); q.act = 2; launch_gemm(stream, q);
      q = plainAB(wo, Wo_, zp, DR * DR, DCZ, CTRI); q.gate = wg; q.residual = 1; launch_gemm(stream, q);
    };
    tri_mul(59, true);   // tri_out (outgoing)
    tri_mul(49, false);  // tri_in  (incoming)

    // ================= Triangle attentions =================
    auto tri_att = [&](int pb, bool starting) {
      const float *Wb=P(pb+0), *Wg_=P(pb+1), *Wk_=P(pb+2), *Wo_=P(pb+3), *Wq_=P(pb+4), *Wv_=P(pb+5),
                  *bg=P(pb+6), *bo=P(pb+7), *ln_b=P(pb+8), *ln_g=P(pb+9);
      GemmP q;
      launch_ln(stream, zp, ln_g, ln_b, wln, DR * DR, DCZ);
      q = plainAB(wln, Wq_, wq, DR * DR, DCZ, DCZ); launch_gemm(stream, q);
      q = plainAB(wln, Wk_, wk, DR * DR, DCZ, DCZ); launch_gemm(stream, q);
      q = plainAB(wln, Wv_, wv, DR * DR, DCZ, DCZ); launch_gemm(stream, q);
      q = plainAB(wln, Wg_, wg, DR * DR, DCZ, DCZ); q.bias = bg; q.act = 2; launch_gemm(stream, q);
      q = plainAB(wln, Wb, wx, DR * DR, HTRI, DCZ); launch_gemm(stream, q);  // bias [R,R,4]
      const long RZ = (long)DR * DCZ;  // 32768
      for (int ic = 0; ic < 8; ++ic) {
        const long I0 = (long)ic * 32;
        if (starting) {  // batch (i,h): rows j, cols k; bias b[h,j,k]
          q = plainAB(wq + I0 * RZ, wk + I0 * RZ, wt, DR, DR, CHD);
          q.a_m1 = DCZ; q.a_k1 = 1; q.a_b1 = RZ; q.a_b2 = CHD;
          q.b_k1 = 1; q.b_n1 = DCZ; q.b_b1 = RZ; q.b_b2 = CHD;
          q.bias = wx; q.s_m = (long)DR * HTRI; q.s_n = HTRI; q.s_b2 = 1;
        } else {         // batch (j,h): rows i, cols k; bias b[h,k,i]
          q = plainAB(wq + I0 * DCZ, wk + I0 * DCZ, wt, DR, DR, CHD);
          q.a_m1 = RZ; q.a_k1 = 1; q.a_b1 = DCZ; q.a_b2 = CHD;
          q.b_k1 = 1; q.b_n1 = RZ; q.b_b1 = DCZ; q.b_b2 = CHD;
          q.bias = wx; q.s_m = HTRI; q.s_n = (long)DR * HTRI; q.s_b2 = 1;
        }
        q.c_m = DR; q.c_n = 1; q.c_b1 = (long)HTRI * DR * DR; q.c_b2 = (long)DR * DR;
        q.alpha = ATT_SCALE; q.b2n = HTRI;
        launch_gemm(stream, q, 32);
        launch_softmax(stream, wt, 32 * HTRI * DR, DR);
        if (starting) {
          q = plainAB(wt, wv + I0 * RZ, wo + I0 * RZ, DR, CHD, DR);
          q.b_k1 = DCZ; q.b_n1 = 1; q.b_b1 = RZ; q.b_b2 = CHD;
          q.c_m = DCZ; q.c_n = 1; q.c_b1 = RZ; q.c_b2 = CHD;
          q.gate = wg + I0 * RZ;
        } else {
          q = plainAB(wt, wv + I0 * DCZ, wo + I0 * DCZ, DR, CHD, DR);
          q.b_k1 = RZ; q.b_n1 = 1; q.b_b1 = DCZ; q.b_b2 = CHD;
          q.c_m = RZ; q.c_n = 1; q.c_b1 = DCZ; q.c_b2 = CHD;
          q.gate = wg + I0 * DCZ;
        }
        q.a_m1 = DR; q.a_k1 = 1; q.a_b1 = (long)HTRI * DR * DR; q.a_b2 = (long)DR * DR;
        q.b2n = HTRI;
        launch_gemm(stream, q, 32);
      }
      q = plainAB(wo, Wo_, zp, DR * DR, DCZ, DCZ); q.bias = bo; q.residual = 1; launch_gemm(stream, q);
    };
    tri_att(69, true);   // tri_start
    tri_att(39, false);  // tri_end

    // ================= Pair transition =================
    {
      const float *W1=P(21), *W2=P(22), *b1=P(23), *b2=P(24), *ln_b=P(25), *ln_g=P(26);
      launch_ln(stream, zp, ln_g, ln_b, wln, DR * DR, DCZ);
      for (int c = 0; c < 4; ++c) {
        const long r0 = (long)c * 16384;
        p = plainAB(wln + r0 * DCZ, W1, wt, 16384, 4 * DCZ, DCZ); p.bias = b1; p.act = 1; launch_gemm(stream, p);
        p = plainAB(wt, W2, zp + r0 * DCZ, 16384, DCZ, 4 * DCZ); p.bias = b2; p.residual = 1; launch_gemm(stream, p);
      }
    }
  }

  // ================= final outputs =================
  launch_ln(stream, zp, in[163], in[162], pairOut, DR * DR, DCZ);   // pair = LN(pair)
  {
    GemmP p = plainAB(msa + (long)DR * DCM, in[2], msas, DR, 128, DCM);  // msa[1] @ Wms + bms
    p.bias = in[3];
    launch_gemm(stream, p);
  }
}
